// USVHeteroGNN_53034256171220
// MI455X (gfx1250) — matrix operand order repaired and checked by reading, not compiled, not run on hardware
//
#include <hip/hip_runtime.h>
#include <hip/hip_bf16.h>
#include <math.h>

// ---------------------------------------------------------------------------
// USV hetero-GNN (2-layer GAT) for gfx1250.
// Dense GEMMs: LDS-staged bf16 tiles -> v_wmma_f32_16x16x32_bf16 (f32 accum).
// Edge softmax: order-independent float atomic-max + atomic-add sums.
// Workspace requirement: ~83 MB (see offsets in kernel_launch).
// ---------------------------------------------------------------------------

#define HID   256
#define HEADS 8
#define FPH   32
#define NT    20000
#define NU    2000
#define E_TT  320000
#define E_UT  100000

typedef __attribute__((ext_vector_type(16))) __bf16    v16bf;
typedef __attribute__((ext_vector_type(2)))  __bf16    v2bf;
typedef __attribute__((ext_vector_type(8)))  float     v8f;
typedef __attribute__((ext_vector_type(8)))  unsigned  v8u;

// ---------------------------------------------------------------------------
// GEMM: C[M,N] = A[M,K] @ B[K,N] (+ bias[N]) (optional relu). f32 in memory,
// bf16 WMMA with f32 accumulation.
//
// Block: 256 threads (8 waves) -> 64x64 C tile. Waves arranged 4x2; each wave
// computes 16 rows x 32 cols = two 16x16 WMMA accumulators sharing one A frag.
// A/B tiles staged through LDS as bf16 (padded stride 34 -> conflict-free
// 4-byte fragment reads); B stored K-major so B fragments are contiguous.
// Requires: N % 64 == 0, K % 32 == 0 (true for all calls: N=256, K in {32,256}).
// M is guarded.
// ---------------------------------------------------------------------------
#define BM 64
#define BN 64
#define BK 32
#define LDK 34   // padded LDS stride in bf16 elements (even, conflict-free)

__device__ __forceinline__ unsigned pack_bf16(float x, float y)
{
    v2bf p = { (__bf16)x, (__bf16)y };
    return __builtin_bit_cast(unsigned, p);
}

__global__ void __launch_bounds__(256)
gemm_wmma_bf16(const float* __restrict__ A,
               const float* __restrict__ B,
               const float* __restrict__ bias,
               float* __restrict__ C,
               int M, int N, int K, int relu)
{
    __shared__ __bf16 a_lds[BM * LDK];
    __shared__ __bf16 b_lds[BN * LDK];

    const int tid  = threadIdx.x;
    const int lane = tid & 31;
    const int wave = tid >> 5;     // 0..7
    const int wm   = wave >> 1;    // 0..3 : 16-row strip
    const int wn   = wave & 1;     // 0..1 : 32-col strip
    const int half = lane >> 4;    // 0 / 1
    const int mr   = lane & 15;

    const int rowBase = blockIdx.y * BM;
    const int colBase = blockIdx.x * BN;

    // cooperative load assignments
    const int ar = tid >> 3;         // A tile row 0..31 (plus +32)
    const int ac = (tid & 7) * 4;    // A tile col (float4)
    const int bk = tid >> 3;         // B tile k 0..31
    const int bc = (tid & 7) * 8;    // B tile col (8 consecutive)

    v8f acc0 = {}, acc1 = {};

    for (int k0 = 0; k0 < K; k0 += BK) {
        // ---- stage A tile (BM x BK) as bf16, row-major padded ----
#pragma unroll
        for (int rr = 0; rr < 2; ++rr) {
            const int r  = ar + rr * 32;
            const int gr = rowBase + r;
            float4 v = make_float4(0.f, 0.f, 0.f, 0.f);
            if (gr < M) v = *(const float4*)(A + (size_t)gr * K + k0 + ac);
            unsigned* dst = (unsigned*)(a_lds + r * LDK + ac);
            dst[0] = pack_bf16(v.x, v.y);
            dst[1] = pack_bf16(v.z, v.w);
        }
        // ---- stage B tile (BK x BN) as bf16, K-major (transposed) ----
        {
            const float* bp = B + (size_t)(k0 + bk) * N + colBase + bc;
            float4 v0 = *(const float4*)(bp);
            float4 v1 = *(const float4*)(bp + 4);
            b_lds[(bc + 0) * LDK + bk] = (__bf16)v0.x;
            b_lds[(bc + 1) * LDK + bk] = (__bf16)v0.y;
            b_lds[(bc + 2) * LDK + bk] = (__bf16)v0.z;
            b_lds[(bc + 3) * LDK + bk] = (__bf16)v0.w;
            b_lds[(bc + 4) * LDK + bk] = (__bf16)v1.x;
            b_lds[(bc + 5) * LDK + bk] = (__bf16)v1.y;
            b_lds[(bc + 6) * LDK + bk] = (__bf16)v1.z;
            b_lds[(bc + 7) * LDK + bk] = (__bf16)v1.w;
        }
        // prefetch next K tile while we compute this one
        if (k0 + BK < K) {
            const int grp = rowBase + ar;
            if (grp < M) __builtin_prefetch(A + (size_t)grp * K + k0 + BK + ac, 0, 3);
            __builtin_prefetch(B + (size_t)(k0 + BK + bk) * N + colBase + bc, 0, 3);
        }
        __syncthreads();

        // ---- build fragments from LDS (all 4-byte aligned reads) ----
        v8u au, bu0, bu1;
        const __bf16* ap  = a_lds + (wm * 16 + mr) * LDK;
        const __bf16* bp0 = b_lds + (wn * 32 + mr) * LDK;
        const __bf16* bp1 = bp0 + 16 * LDK;
#pragma unroll
        for (int i = 0; i < 8; ++i) {
            // A 16x32 bf16 layout: lanes 0-15 hold K {0..7,16..23},
            // lanes 16-31 hold K {8..15,24..31}, 2 values per VGPR.
            const int ka = ((i & 4) ? 16 : 0) + half * 8 + 2 * (i & 3);
            au[i] = *(const unsigned*)(ap + ka);
            // B 32x16 bf16 layout: same K order as A (lanes 0-15 K {0..7,16..23}, lanes 16-31 K {8..15,24..31}).
            const int kb = ((i & 4) ? 16 : 0) + half * 8 + 2 * (i & 3);
            bu0[i] = *(const unsigned*)(bp0 + kb);
            bu1[i] = *(const unsigned*)(bp1 + kb);
        }
        const v16bf av = __builtin_bit_cast(v16bf, au);
        acc0 = __builtin_amdgcn_wmma_f32_16x16x32_bf16(
            false, av, false, __builtin_bit_cast(v16bf, bu0), (short)0, acc0, false, false);
        acc1 = __builtin_amdgcn_wmma_f32_16x16x32_bf16(
            false, av, false, __builtin_bit_cast(v16bf, bu1), (short)0, acc1, false, false);
        __syncthreads();
    }

    // ---- writeback: lanes 0-15 -> rows r (N=lane), lanes 16-31 -> rows r+8 ----
    const int row0 = rowBase + wm * 16 + half * 8;
    const int cc0  = colBase + wn * 32 + mr;
    const int cc1  = cc0 + 16;
    const float b0 = bias ? bias[cc0] : 0.0f;
    const float b1 = bias ? bias[cc1] : 0.0f;
#pragma unroll
    for (int r = 0; r < 8; ++r) {
        const int row = row0 + r;
        if (row >= M) break;
        float x0 = acc0[r] + b0;
        float x1 = acc1[r] + b1;
        if (relu) { x0 = fmaxf(x0, 0.f); x1 = fmaxf(x1, 0.f); }
        C[(size_t)row * N + cc0] = x0;
        C[(size_t)row * N + cc1] = x1;
    }
}

// ---------------------------------------------------------------------------
// Small helper kernels
// ---------------------------------------------------------------------------
__global__ void fill_f32(float* p, float v, long n)
{
    long i = (long)blockIdx.x * blockDim.x + threadIdx.x;
    if (i < n) p[i] = v;
}

// el[n,h] = sum_f X[n, h*32+f] * attn[h*32+f]
__global__ void head_logit(const float* __restrict__ X,
                           const float* __restrict__ attn,
                           float* __restrict__ out, int n_nodes)
{
    int gid = blockIdx.x * blockDim.x + threadIdx.x;
    if (gid >= n_nodes * HEADS) return;
    int n = gid >> 3, h = gid & 7;
    const float* x = X + (size_t)n * HID + h * FPH;
    const float* a = attn + h * FPH;
    float acc = 0.f;
#pragma unroll
    for (int f = 0; f < FPH; ++f) acc += x[f] * a[f];
    out[gid] = acc;
}

__device__ __forceinline__ void atomic_max_f32(float* addr, float val)
{
    // Order-independent (exactly deterministic) float max via int punning.
    if (val >= 0.0f) atomicMax((int*)addr, __float_as_int(val));
    else             atomicMin((unsigned int*)addr, (unsigned int)__float_as_int(val));
}

// z = leaky_relu(el[src]+er[dst]); store z; segment-max into m[dst,h]
__global__ void edge_logit_max(const int* __restrict__ src, const int* __restrict__ dst,
                               const float* __restrict__ el, const float* __restrict__ er,
                               float* __restrict__ ez, float* __restrict__ m, int E)
{
    int gid = blockIdx.x * blockDim.x + threadIdx.x;
    if (gid >= E * HEADS) return;
    int e = gid >> 3, h = gid & 7;
    float z = el[src[e] * HEADS + h] + er[dst[e] * HEADS + h];
    z = (z < 0.f) ? 0.2f * z : z;
    ez[gid] = z;
    atomic_max_f32(&m[dst[e] * HEADS + h], z);
}

// w = exp(z - m[dst,h]) (in place); segment-sum into s[dst,h]
__global__ void edge_exp_sum(const int* __restrict__ dst,
                             const float* __restrict__ m, float* __restrict__ ez,
                             float* __restrict__ s, int E)
{
    int gid = blockIdx.x * blockDim.x + threadIdx.x;
    if (gid >= E * HEADS) return;
    int e = gid >> 3, h = gid & 7;
    float w = __expf(ez[gid] - m[dst[e] * HEADS + h]);
    ez[gid] = w;
    atomicAdd(&s[dst[e] * HEADS + h], w);
}

// out[dst, j] += (w[e,h]/s[dst,h]) * fs[src, j],  j = h*32+f
__global__ void edge_aggregate(const int* __restrict__ src, const int* __restrict__ dst,
                               const float* __restrict__ fs, const float* __restrict__ ew,
                               const float* __restrict__ s, float* __restrict__ out, int E)
{
    size_t gid = (size_t)blockIdx.x * blockDim.x + threadIdx.x;
    if (gid >= (size_t)E * HID) return;
    int e = (int)(gid >> 8);
    int j = (int)(gid & 255);
    int h = j >> 5;
    int d = dst[e];
    float a = ew[(size_t)e * HEADS + h] / s[(size_t)d * HEADS + h];
    atomicAdd(&out[(size_t)d * HID + j], a * fs[(size_t)src[e] * HID + j]);
}

// hm[n,f] = mean_h( out[n,h*32+f] + b1[h*32+f] (+ b2[h*32+f]) )
__global__ void head_mean(const float* __restrict__ out,
                          const float* __restrict__ b1, const float* __restrict__ b2,
                          float* __restrict__ hm, int n_nodes)
{
    int gid = blockIdx.x * blockDim.x + threadIdx.x;
    if (gid >= n_nodes * FPH) return;
    int n = gid >> 5, f = gid & 31;
    float acc = 0.f;
#pragma unroll
    for (int h = 0; h < HEADS; ++h) {
        float b = b1[h * FPH + f] + (b2 ? b2[h * FPH + f] : 0.f);
        acc += out[(size_t)n * HID + h * FPH + f] + b;
    }
    hm[gid] = acc * 0.125f;
}

// column sums of X[M,256] accumulated into out[256] (out pre-zeroed)
__global__ void col_sum(const float* __restrict__ X, float* __restrict__ out, int M)
{
    int col = threadIdx.x;           // 256 threads
    float acc = 0.f;
    for (int r = blockIdx.x; r < M; r += gridDim.x)
        acc += X[(size_t)r * HID + col];
    atomicAdd(&out[col], acc);
}

// global_emb[j] = sum_i (gsum[i]/count_i) * gp_w[i*512+j] + gp_b[j]
__global__ void global_head(const float* __restrict__ gsum,
                            const float* __restrict__ gp_w,
                            const float* __restrict__ gp_b,
                            float* __restrict__ out)
{
    int j = blockIdx.x * blockDim.x + threadIdx.x;
    if (j >= 2 * HID) return;
    float acc = gp_b[j];
    for (int i = 0; i < 2 * HID; ++i) {
        float xi = gsum[i] * (i < HID ? (1.0f / NU) : (1.0f / NT));
        acc += xi * gp_w[(size_t)i * (2 * HID) + j];
    }
    out[j] = acc;
}

// ---------------------------------------------------------------------------
extern "C" void kernel_launch(void* const* d_in, const int* in_sizes, int n_in,
                              void* d_out, int out_size, void* d_ws, size_t ws_size,
                              hipStream_t stream)
{
    const float* usv_feat   = (const float*)d_in[0];
    const float* task_feat  = (const float*)d_in[1];
    const float* usv_enc_w  = (const float*)d_in[2];
    const float* usv_enc_b  = (const float*)d_in[3];
    const float* task_enc_w = (const float*)d_in[4];
    const float* task_enc_b = (const float*)d_in[5];
    const float* l0_tt_wsrc = (const float*)d_in[6];
    const float* l0_tt_wdst = (const float*)d_in[7];
    const float* l0_tt_al   = (const float*)d_in[8];
    const float* l0_tt_ar   = (const float*)d_in[9];
    const float* l0_tt_bias = (const float*)d_in[10];
    const float* l0_ut_wsrc = (const float*)d_in[11];
    const float* l0_ut_wdst = (const float*)d_in[12];
    const float* l0_ut_al   = (const float*)d_in[13];
    const float* l0_ut_ar   = (const float*)d_in[14];
    const float* l0_ut_bias = (const float*)d_in[15];
    const float* l0_post_w  = (const float*)d_in[16];
    const float* l0_post_b  = (const float*)d_in[17];
    const float* l1_tt_wsrc = (const float*)d_in[18];
    const float* l1_tt_wdst = (const float*)d_in[19];
    const float* l1_tt_al   = (const float*)d_in[20];
    const float* l1_tt_ar   = (const float*)d_in[21];
    const float* l1_tt_bias = (const float*)d_in[22];
    const float* l1_post_w  = (const float*)d_in[23];
    const float* l1_post_b  = (const float*)d_in[24];
    const float* task_dec_w = (const float*)d_in[25];
    const float* task_dec_b = (const float*)d_in[26];
    const float* gp_w       = (const float*)d_in[27];
    const float* gp_b       = (const float*)d_in[28];
    const int*   tt_src     = (const int*)d_in[29];
    const int*   tt_dst     = (const int*)d_in[30];
    const int*   ut_src     = (const int*)d_in[31];
    const int*   ut_dst     = (const int*)d_in[32];

    float* out = (float*)d_out;
    float* usv_emb  = out;                          // [NU,256]
    float* task_emb = out + (size_t)NU * HID;       // [NT,256]
    float* glob     = task_emb + (size_t)NT * HID;  // [512]

    // ---- workspace layout (floats); total ~20.69M floats (~83 MB) ----
    float* ws  = (float*)d_ws;
    float* W0  = ws;                         // task_h (current layer)  [NT,256]
    float* W1  = W0  + (size_t)NT * HID;     // fs_tt                   [NT,256]
    float* W2  = W1  + (size_t)NT * HID;     // fd scratch -> out accum [NT,256]
    float* FSU = W2  + (size_t)NT * HID;     // fs_ut                   [NU,256]
    float* EL  = FSU + (size_t)NU * HID;     // el (task-level)         [NT,8]
    float* ER  = EL  + (size_t)NT * HEADS;   // er_tt                   [NT,8]
    float* ELU = ER  + (size_t)NT * HEADS;   // el_ut (usv-level)       [NU,8]
    float* ERU = ELU + (size_t)NU * HEADS;   // er_ut                   [NT,8]
    float* Mx  = ERU + (size_t)NT * HEADS;   // segment max             [NT,8]
    float* Sx  = Mx  + (size_t)NT * HEADS;   // segment sum             [NT,8]
    float* EZT = Sx  + (size_t)NT * HEADS;   // edge buffer tt          [E_TT,8]
    float* EZU = EZT + (size_t)E_TT * HEADS; // edge buffer ut          [E_UT,8]
    float* HM  = EZU + (size_t)E_UT * HEADS; // head-mean               [NT,32]
    float* GS  = HM  + (size_t)NT * FPH;     // global pooled sums      [512]

    const float NEG_INF = -__builtin_huge_valf();

    auto gemm = [&](const float* A, const float* B, const float* bias, float* C,
                    int M, int N, int K, int relu) {
        dim3 grid(N / BN, (M + BM - 1) / BM);
        gemm_wmma_bf16<<<grid, 256, 0, stream>>>(A, B, bias, C, M, N, K, relu);
    };
    auto fill = [&](float* p, float v, long n) {
        fill_f32<<<(unsigned)((n + 255) / 256), 256, 0, stream>>>(p, v, n);
    };
    auto gat_edges = [&](const int* src, const int* dst, int E,
                         const float* el, const float* er,
                         const float* fs, float* ez, float* outbuf) {
        fill(Mx, NEG_INF, (long)NT * HEADS);
        fill(Sx, 0.0f,    (long)NT * HEADS);
        int gE8 = (E * HEADS + 255) / 256;
        edge_logit_max<<<gE8, 256, 0, stream>>>(src, dst, el, er, ez, Mx, E);
        edge_exp_sum  <<<gE8, 256, 0, stream>>>(dst, Mx, ez, Sx, E);
        unsigned gE256 = (unsigned)(((size_t)E * HID + 255) / 256);
        edge_aggregate<<<gE256, 256, 0, stream>>>(src, dst, fs, ez, Sx, outbuf, E);
    };

    // ===== Encoders =====
    gemm(usv_feat,  usv_enc_w,  usv_enc_b,  usv_emb, NU, HID, 32, 0); // usv_h == usv_emb
    gemm(task_feat, task_enc_w, task_enc_b, W0,      NT, HID, 32, 0); // task_h

    // ===== Layer 0 =====
    gemm(W0, l0_tt_wsrc, nullptr, W1, NT, HID, HID, 0);               // fs_tt
    head_logit<<<(NT * HEADS + 255) / 256, 256, 0, stream>>>(W1, l0_tt_al, EL, NT);
    gemm(W0, l0_tt_wdst, nullptr, W2, NT, HID, HID, 0);               // fd_tt
    head_logit<<<(NT * HEADS + 255) / 256, 256, 0, stream>>>(W2, l0_tt_ar, ER, NT);
    gemm(W0, l0_ut_wdst, nullptr, W2, NT, HID, HID, 0);               // fd_ut (reuse W2)
    head_logit<<<(NT * HEADS + 255) / 256, 256, 0, stream>>>(W2, l0_ut_ar, ERU, NT);
    gemm(usv_emb, l0_ut_wsrc, nullptr, FSU, NU, HID, HID, 0);         // fs_ut
    head_logit<<<(NU * HEADS + 255) / 256, 256, 0, stream>>>(FSU, l0_ut_al, ELU, NU);

    fill(W2, 0.0f, (long)NT * HID);                                   // out accumulator
    gat_edges(tt_src, tt_dst, E_TT, EL,  ER,  W1,  EZT, W2);          // task->task
    gat_edges(ut_src, ut_dst, E_UT, ELU, ERU, FSU, EZU, W2);          // usv->task (summed)

    head_mean<<<(NT * FPH + 255) / 256, 256, 0, stream>>>(W2, l0_tt_bias, l0_ut_bias, HM, NT);
    gemm(HM, l0_post_w, l0_post_b, W0, NT, HID, 32, 0);               // task_h (layer 1 input)

    // ===== Layer 1 (tt only) =====
    gemm(W0, l1_tt_wsrc, nullptr, W1, NT, HID, HID, 0);               // fs
    head_logit<<<(NT * HEADS + 255) / 256, 256, 0, stream>>>(W1, l1_tt_al, EL, NT);
    gemm(W0, l1_tt_wdst, nullptr, W2, NT, HID, HID, 0);               // fd
    head_logit<<<(NT * HEADS + 255) / 256, 256, 0, stream>>>(W2, l1_tt_ar, ER, NT);

    fill(W2, 0.0f, (long)NT * HID);
    gat_edges(tt_src, tt_dst, E_TT, EL, ER, W1, EZT, W2);

    head_mean<<<(NT * FPH + 255) / 256, 256, 0, stream>>>(W2, l1_tt_bias, nullptr, HM, NT);
    gemm(HM, l1_post_w, l1_post_b, W0, NT, HID, 32, 0);               // task_h final

    // ===== Decoder + global pooling =====
    gemm(W0, task_dec_w, task_dec_b, task_emb, NT, HID, HID, 1);      // relu

    fill(GS, 0.0f, 2 * HID);
    col_sum<<<64, HID, 0, stream>>>(usv_emb,  GS,       NU);
    col_sum<<<64, HID, 0, stream>>>(task_emb, GS + HID, NT);
    global_head<<<2, HID, 0, stream>>>(GS, gp_w, gp_b, glob);

    (void)in_sizes; (void)n_in; (void)out_size; (void)ws_size;
}
